// SkeletonGCN_33243046871577
// MI455X (gfx1250) — matrix operand order repaired and checked by reading, not compiled, not run on hardware
//
#include <hip/hip_runtime.h>
#include <hip/hip_bf16.h>
#include <math.h>

// ---------------------------------------------------------------------------
// GCN pipeline for MI455X (gfx1250, wave32).
// GEMMs use v_wmma_f32_16x16x32_bf16 (bf16 in, f32 accumulate), with W staged
// to LDS pre-swizzled into WMMA B-fragment order (wide ds_load per fragment).
// Aggregation = L2-resident f32 atomics (dominant cost; bandwidth/atomic bound).
// ---------------------------------------------------------------------------

#define H64  64
#define CCLS 10
#define EPS_BN 1e-5f

typedef __attribute__((ext_vector_type(16))) __bf16 v16bf;
typedef __attribute__((ext_vector_type(8)))  float  v8f;

__device__ __forceinline__ __bf16 to_bf(float x) { return (__bf16)x; }

// ---------------------------------------------------------------------------
// Generic zero
__global__ void zero_f32(float* __restrict__ p, int n) {
    int i = blockIdx.x * blockDim.x + threadIdx.x;
    if (i < n) p[i] = 0.0f;
}

// ---------------------------------------------------------------------------
// Degree with self loop, then dinv = rsqrt(max(deg,1))
__global__ void deg_init(float* __restrict__ deg, int n) {
    int i = blockIdx.x * blockDim.x + threadIdx.x;
    if (i < n) deg[i] = 1.0f;                    // self loop
}
__global__ void deg_accum(const int* __restrict__ dst, float* __restrict__ deg, int e) {
    int i = blockIdx.x * blockDim.x + threadIdx.x;
    if (i < e) atomicAdd(&deg[dst[i]], 1.0f);
}
__global__ void deg_to_dinv(float* __restrict__ deg, int n) {
    int i = blockIdx.x * blockDim.x + threadIdx.x;
    if (i < n) deg[i] = rsqrtf(fmaxf(deg[i], 1.0f));
}
__global__ void edge_norm(const int* __restrict__ src, const int* __restrict__ dst,
                          const float* __restrict__ dinv, float* __restrict__ nrm, int e) {
    int i = blockIdx.x * blockDim.x + threadIdx.x;
    if (i < e) nrm[i] = dinv[src[i]] * dinv[dst[i]];
}

// ---------------------------------------------------------------------------
// C[nrows x 64] = A[nrows x 64] @ W[64 x 64]  via WMMA bf16, f32 accumulate.
//
// Block = 256 threads = 8 waves; each wave owns 64 rows (4 x 16-row tiles).
// W is staged to LDS as bf16 in *fragment order*: fragment f = kstep*4 + nt
// occupies wlds[f*512 + lane*16 + e], e = 0..15, so that each lane loads its
// whole 32-byte B fragment contiguously (wide ds loads, no 16-bit assembly).
//   B frag element e (lane: half = lane>>4, m = lane&15):
//       W[kstep*32 + (e>>3)*16 + half*8 + (e&7)][nt*16 + m]
__global__ void __launch_bounds__(256)
gemm_bf16_n64(const float* __restrict__ A, const float* __restrict__ W,
              float* __restrict__ C, int nrows)
{
    __shared__ __bf16 wlds[8 * 512];   // 8 fragments x (32 lanes x 16 elems) = 8KB

    // swizzle-stage W (row-major 64x64 f32) into fragment order as bf16
    for (int i = threadIdx.x; i < 64 * 64; i += 256) {
        const int k = i >> 6, col = i & 63;
        const int kstep = k >> 5, kk = k & 31;
        const int half = (kk >> 3) & 1, e = (kk & 7) | ((kk >> 4) << 3);
        const int nt = col >> 4, m = col & 15;
        wlds[(kstep * 4 + nt) * 512 + (half * 16 + m) * 16 + e] = to_bf(W[i]);
    }
    __syncthreads();

    const int lane = threadIdx.x & 31;
    const int wave = threadIdx.x >> 5;
    const int m    = lane & 15;
    const int half = lane >> 4;

    // 8 B fragments, reused for all 4 row tiles of this wave
    v16bf bfrag[8];
#pragma unroll
    for (int f = 0; f < 8; ++f)
        bfrag[f] = *(const v16bf*)&wlds[f * 512 + lane * 16];

    // A fragment (16-bit A 16x32 layout):
    // lane(half,m): elems 0..7  <- A[row][kofs + half*8 + 0..7]
    //               elems 8..15 <- A[row][kofs + 16 + half*8 + 0..7]
    auto packA = [&](const float* base) -> v16bf {
        float4 p0 = *(const float4*)(base + 0);
        float4 p1 = *(const float4*)(base + 4);
        float4 p2 = *(const float4*)(base + 16);
        float4 p3 = *(const float4*)(base + 20);
        v16bf r;
        r[0] = to_bf(p0.x);  r[1] = to_bf(p0.y);  r[2] = to_bf(p0.z);  r[3] = to_bf(p0.w);
        r[4] = to_bf(p1.x);  r[5] = to_bf(p1.y);  r[6] = to_bf(p1.z);  r[7] = to_bf(p1.w);
        r[8] = to_bf(p2.x);  r[9] = to_bf(p2.y);  r[10]= to_bf(p2.z);  r[11]= to_bf(p2.w);
        r[12]= to_bf(p3.x);  r[13]= to_bf(p3.y);  r[14]= to_bf(p3.z);  r[15]= to_bf(p3.w);
        return r;
    };

    const int waveBase = blockIdx.x * 512 + wave * 64;

#pragma unroll 1
    for (int t = 0; t < 4; ++t) {
        const int row0 = waveBase + t * 16;
        if (row0 >= nrows) break;
        const int row = row0 + m;

        v16bf aLo = {}, aHi = {};
        if (row < nrows) {
            const float* ar = A + (size_t)row * H64;
            aLo = packA(ar + half * 8);        // K 0..31
            aHi = packA(ar + 32 + half * 8);   // K 32..63
        }

        v8f zacc = {};
        v8f acc[4];
#pragma unroll
        for (int nt = 0; nt < 4; ++nt) acc[nt] = zacc;

#pragma unroll
        for (int nt = 0; nt < 4; ++nt) {
            acc[nt] = __builtin_amdgcn_wmma_f32_16x16x32_bf16(
                false, aLo, false, bfrag[nt], (short)0, acc[nt], false, false);
            acc[nt] = __builtin_amdgcn_wmma_f32_16x16x32_bf16(
                false, aHi, false, bfrag[4 + nt], (short)0, acc[nt], false, false);
        }

        // C/D layout: elem r -> row0 + half*8 + r, col nt*16 + m
        float* cb = C + (size_t)(row0 + half * 8) * H64 + m;
        if (row0 + 16 <= nrows) {
            // fast path: unguarded immediate-offset stores
#pragma unroll
            for (int nt = 0; nt < 4; ++nt)
#pragma unroll
                for (int r = 0; r < 8; ++r)
                    cb[(size_t)r * H64 + nt * 16] = acc[nt][r];
        } else {
#pragma unroll
            for (int r = 0; r < 8; ++r) {
                if (row0 + half * 8 + r < nrows) {
#pragma unroll
                    for (int nt = 0; nt < 4; ++nt)
                        cb[(size_t)r * H64 + nt * 16] = acc[nt][r];
                }
            }
        }
    }
}

// ---------------------------------------------------------------------------
// agg = tmp * dinv[n]^2   (self-loop contribution doubles as zero-init)
__global__ void agg_init(const float* __restrict__ tmp, const float* __restrict__ dinv,
                         float* __restrict__ agg, int nrows)
{
    int t = blockIdx.x * blockDim.x + threadIdx.x;
    int total = nrows * 16;
    if (t >= total) return;
    int n = t >> 4, q = t & 15;
    float d = dinv[n];
    float s = d * d;
    float4 v = *(const float4*)(tmp + (size_t)n * H64 + q * 4);
    v.x *= s; v.y *= s; v.z *= s; v.w *= s;
    *(float4*)(agg + (size_t)n * H64 + q * 4) = v;
}

// agg[dst] += tmp[src] * norm[e] ; 16 threads / edge, float4 each
__global__ void edge_scatter(const int* __restrict__ src, const int* __restrict__ dst,
                             const float* __restrict__ nrm,
                             const float* __restrict__ tmp, float* __restrict__ agg, int e)
{
    long long t = (long long)blockIdx.x * blockDim.x + threadIdx.x;
    long long total = (long long)e * 16;
    if (t >= total) return;
    int ei = (int)(t >> 4), q = (int)(t & 15);
    int s = src[ei], d = dst[ei];
    float w = nrm[ei];
    float4 v = *(const float4*)(tmp + (size_t)s * H64 + q * 4);
    float* o = agg + (size_t)d * H64 + q * 4;
    atomicAdd(o + 0, v.x * w);
    atomicAdd(o + 1, v.y * w);
    atomicAdd(o + 2, v.z * w);
    atomicAdd(o + 3, v.w * w);
}

// ---------------------------------------------------------------------------
// Per-channel sum / sumsq of (h + bias)
__global__ void __launch_bounds__(256)
bn_stats(const float* __restrict__ h, const float* __restrict__ bias,
         float* __restrict__ sum, float* __restrict__ sumsq, int nrows)
{
    __shared__ float s0[256], s1[256];
    const int c   = threadIdx.x & 63;
    const int sub = threadIdx.x >> 6;   // 0..3
    float b = bias[c];
    float a = 0.0f, a2 = 0.0f;
    for (int row = blockIdx.x * 4 + sub; row < nrows; row += gridDim.x * 4) {
        float v = h[(size_t)row * H64 + c] + b;
        a += v; a2 += v * v;
    }
    s0[threadIdx.x] = a;
    s1[threadIdx.x] = a2;
    __syncthreads();
    if (threadIdx.x < 64) {
        float r  = s0[threadIdx.x] + s0[threadIdx.x + 64] + s0[threadIdx.x + 128] + s0[threadIdx.x + 192];
        float r2 = s1[threadIdx.x] + s1[threadIdx.x + 64] + s1[threadIdx.x + 128] + s1[threadIdx.x + 192];
        atomicAdd(&sum[threadIdx.x],   r);
        atomicAdd(&sumsq[threadIdx.x], r2);
    }
}

// h = relu(((h+bias)-mean)*rsqrt(var+eps)*g + be) [+ residual], in place
__global__ void bn_apply(float* __restrict__ h, const float* __restrict__ bias,
                         const float* __restrict__ sum, const float* __restrict__ sumsq,
                         const float* __restrict__ gamma, const float* __restrict__ beta,
                         const float* __restrict__ residual, int nrows)
{
    int i = blockIdx.x * blockDim.x + threadIdx.x;
    int total = nrows * H64;
    if (i >= total) return;
    int c = i & 63;
    float invN = 1.0f / (float)nrows;
    float mean = sum[c] * invN;
    float var  = sumsq[c] * invN - mean * mean;
    float v = h[i] + bias[c];
    v = (v - mean) * rsqrtf(var + EPS_BN) * gamma[c] + beta[c];
    v = fmaxf(v, 0.0f);
    if (residual) v += residual[i];
    h[i] = v;
}

// ---------------------------------------------------------------------------
// Global mean pool
__global__ void pool_count(const int* __restrict__ batch, float* __restrict__ cnt, int nrows) {
    int n = blockIdx.x * blockDim.x + threadIdx.x;
    if (n < nrows) atomicAdd(&cnt[batch[n]], 1.0f);
}
__global__ void pool_sum(const float* __restrict__ h, const int* __restrict__ batch,
                         float* __restrict__ pool, int nrows)
{
    int t = blockIdx.x * blockDim.x + threadIdx.x;
    int total = nrows * 16;
    if (t >= total) return;
    int n = t >> 4, q = t & 15;
    float4 v = *(const float4*)(h + (size_t)n * H64 + q * 4);
    float* o = pool + (size_t)batch[n] * H64 + q * 4;
    atomicAdd(o + 0, v.x);
    atomicAdd(o + 1, v.y);
    atomicAdd(o + 2, v.z);
    atomicAdd(o + 3, v.w);
}

// ---------------------------------------------------------------------------
// Head: pooled -> relu(64x32) -> 32x10 -> log_softmax. One 64-thread block/graph.
__global__ void __launch_bounds__(64)
head_kernel(const float* __restrict__ pool, const float* __restrict__ cnt,
            const float* __restrict__ lw1, const float* __restrict__ lb1,
            const float* __restrict__ lw2, const float* __restrict__ lb2,
            float* __restrict__ out)
{
    __shared__ float p[64];
    __shared__ float z[32];
    __shared__ float lg[CCLS];
    __shared__ float red[2];
    const int g = blockIdx.x;
    const int t = threadIdx.x;

    float inv = 1.0f / fmaxf(cnt[g], 1.0f);
    p[t] = pool[(size_t)g * H64 + t] * inv;
    __syncthreads();

    if (t < 32) {
        float a = lb1[t];
#pragma unroll
        for (int k = 0; k < 64; ++k) a += p[k] * lw1[k * 32 + t];
        z[t] = fmaxf(a, 0.0f);
    }
    __syncthreads();

    if (t < CCLS) {
        float a = lb2[t];
#pragma unroll
        for (int j = 0; j < 32; ++j) a += z[j] * lw2[j * CCLS + t];
        lg[t] = a;
    }
    __syncthreads();

    if (t == 0) {
        float mx = lg[0];
        for (int i = 1; i < CCLS; ++i) mx = fmaxf(mx, lg[i]);
        float se = 0.0f;
        for (int i = 0; i < CCLS; ++i) se += expf(lg[i] - mx);
        red[0] = mx;
        red[1] = logf(se);
    }
    __syncthreads();

    if (t < CCLS) out[(size_t)g * CCLS + t] = lg[t] - red[0] - red[1];
}

// ---------------------------------------------------------------------------
extern "C" void kernel_launch(void* const* d_in, const int* in_sizes, int n_in,
                              void* d_out, int out_size, void* d_ws, size_t ws_size,
                              hipStream_t stream)
{
    const float* x   = (const float*)d_in[0];
    const int*   ei  = (const int*)  d_in[1];
    const int*   bat = (const int*)  d_in[2];
    const float* W1  = (const float*)d_in[3];
    const float* b1  = (const float*)d_in[4];
    const float* g1  = (const float*)d_in[5];
    const float* be1 = (const float*)d_in[6];
    const float* W2  = (const float*)d_in[7];
    const float* b2  = (const float*)d_in[8];
    const float* g2  = (const float*)d_in[9];
    const float* be2 = (const float*)d_in[10];
    const float* lw1 = (const float*)d_in[11];
    const float* lb1 = (const float*)d_in[12];
    const float* lw2 = (const float*)d_in[13];
    const float* lb2 = (const float*)d_in[14];
    float* out = (float*)d_out;
    (void)n_in; (void)ws_size;

    const int N = in_sizes[0] / H64;
    const int E = in_sizes[1] / 2;
    const int G = out_size / CCLS;
    const int* src = ei;
    const int* dst = ei + E;

    // workspace carve-out (256B aligned slabs)
    char* ws = (char*)d_ws;
    size_t off = 0;
    auto carve = [&](size_t bytes) -> float* {
        float* p = (float*)(ws + off);
        off += (bytes + 255) & ~(size_t)255;
        return p;
    };
    float* dinv  = carve((size_t)N * 4);            // degree then dinv (in place)
    float* nrm   = carve((size_t)E * 4);            // per-edge norm
    float* bufA  = carve((size_t)N * H64 * 4);      // GEMM outputs (tmp1/tmp2)
    float* bufB  = carve((size_t)N * H64 * 4);      // agg1 -> h1
    float* bufC  = carve((size_t)N * H64 * 4);      // agg2 -> h2
    float* bnsum = carve(64 * 4);
    float* bnsq  = carve(64 * 4);
    float* pools = carve((size_t)G * H64 * 4);
    float* cnt   = carve((size_t)G * 4);

    const int TB = 256;
    auto cdiv = [](long long a, long long b) -> unsigned int { return (unsigned int)((a + b - 1) / b); };

    // degrees + norms
    deg_init   <<<cdiv(N, TB), TB, 0, stream>>>(dinv, N);
    deg_accum  <<<cdiv(E, TB), TB, 0, stream>>>(dst, dinv, E);
    deg_to_dinv<<<cdiv(N, TB), TB, 0, stream>>>(dinv, N);
    edge_norm  <<<cdiv(E, TB), TB, 0, stream>>>(src, dst, dinv, nrm, E);

    const unsigned int gemmBlocks = cdiv(N, 512);   // 512 rows per block (8 waves x 64 rows)

    // ---- conv1 ----
    gemm_bf16_n64<<<gemmBlocks, TB, 0, stream>>>(x, W1, bufA, N);
    agg_init     <<<cdiv((long long)N * 16, TB), TB, 0, stream>>>(bufA, dinv, bufB, N);
    edge_scatter <<<cdiv((long long)E * 16, TB), TB, 0, stream>>>(src, dst, nrm, bufA, bufB, E);
    zero_f32     <<<1, 64, 0, stream>>>(bnsum, 64);
    zero_f32     <<<1, 64, 0, stream>>>(bnsq, 64);
    bn_stats     <<<512, TB, 0, stream>>>(bufB, b1, bnsum, bnsq, N);
    bn_apply     <<<cdiv((long long)N * H64, TB), TB, 0, stream>>>(bufB, b1, bnsum, bnsq, g1, be1, nullptr, N);

    // ---- conv2 (+ residual h1 in bufB) ----
    gemm_bf16_n64<<<gemmBlocks, TB, 0, stream>>>(bufB, W2, bufA, N);
    agg_init     <<<cdiv((long long)N * 16, TB), TB, 0, stream>>>(bufA, dinv, bufC, N);
    edge_scatter <<<cdiv((long long)E * 16, TB), TB, 0, stream>>>(src, dst, nrm, bufA, bufC, E);
    zero_f32     <<<1, 64, 0, stream>>>(bnsum, 64);
    zero_f32     <<<1, 64, 0, stream>>>(bnsq, 64);
    bn_stats     <<<512, TB, 0, stream>>>(bufC, b2, bnsum, bnsq, N);
    bn_apply     <<<cdiv((long long)N * H64, TB), TB, 0, stream>>>(bufC, b2, bnsum, bnsq, g2, be2, bufB, N);

    // ---- pool + head ----
    zero_f32  <<<cdiv((long long)G * H64, TB), TB, 0, stream>>>(pools, G * H64);
    zero_f32  <<<cdiv(G, TB), TB, 0, stream>>>(cnt, G);
    pool_count<<<cdiv(N, TB), TB, 0, stream>>>(bat, cnt, N);
    pool_sum  <<<cdiv((long long)N * 16, TB), TB, 0, stream>>>(bufC, bat, pools, N);
    head_kernel<<<G, 64, 0, stream>>>(pools, cnt, lw1, lb1, lw2, lb2, out);
}
